// RewardPredModel_30940944400567
// MI455X (gfx1250) — matrix operand order repaired and checked by reading, not compiled, not run on hardware
//
#include <hip/hip_runtime.h>

typedef __bf16 bf16;
typedef __attribute__((ext_vector_type(16))) __bf16 v16bf;
typedef __attribute__((ext_vector_type(8)))  float  v8f;

#define WAVE 32
#define WPB  8            // waves per 256-thread block
#define TPB  (WAVE*WPB)

#if defined(__has_builtin)
# if __has_builtin(__builtin_amdgcn_tensor_load_to_lds)
#  define HAVE_TDM 1
# endif
#endif
#ifndef HAVE_TDM
# define HAVE_TDM 0
#endif

__device__ __forceinline__ float ldf(const float* p) { return *p; }
__device__ __forceinline__ float ldf(const bf16*  p) { return (float)*p; }

static __device__ __forceinline__ v8f wmma_bf16(v16bf a, v16bf b, v8f c) {
  return __builtin_amdgcn_wmma_f32_16x16x32_bf16(false, a, false, b, (short)0, c,
                                                 false, false);
}

#if HAVE_TDM
typedef __attribute__((ext_vector_type(4))) unsigned tdm_v4u;
typedef __attribute__((ext_vector_type(8))) int      tdm_v8i;
typedef __attribute__((ext_vector_type(4))) int      tdm_v4i;

// 1-D fp32 TDM load: tile == tensor, nElems fp32 words, global -> LDS.
// D# packing per CDNA5 ISA.
__device__ __forceinline__ void tdm_load_f32(void* ldsDst, const float* g,
                                             unsigned nElems) {
  unsigned long long ga = (unsigned long long)(size_t)g;
  unsigned ldsOff = (unsigned)(size_t)ldsDst;          // low 32b = LDS offset
  tdm_v4u g0;
  g0[0] = 1u;                                          // count=1 (valid)
  g0[1] = ldsOff;                                      // lds_addr (bytes)
  g0[2] = (unsigned)(ga & 0xFFFFFFFFu);                // global_addr[31:0]
  g0[3] = (unsigned)((ga >> 32) & 0x1FFFFFFu)          // global_addr[56:32]
        | (2u << 30);                                  // type = 2 ("image")
  tdm_v8i g1;
  g1[0] = (int)(2u << 16);                             // wg_mask=0, data_size=4B
  g1[1] = (int)((nElems & 0xFFFFu) << 16);             // tensor_dim0[15:0]
  g1[2] = (int)(((nElems >> 16) & 0xFFFFu)             // tensor_dim0[31:16]
        | (1u << 16));                                 // tensor_dim1 = 1
  g1[3] = (int)((nElems & 0xFFFFu) << 16);             // tile_dim0 = nElems
  g1[4] = 1;                                           // tile_dim1=1, tile_dim2=0
  g1[5] = (int)nElems;                                 // tensor_dim0_stride[31:0]
  g1[6] = 0;                                           // stride hi / dim1_stride
  g1[7] = 0;
  tdm_v4i z = {0, 0, 0, 0};
#if __clang_major__ >= 23
  tdm_v8i z8 = {0, 0, 0, 0, 0, 0, 0, 0};
  __builtin_amdgcn_tensor_load_to_lds(g0, g1, z, z, z8, 0);
#else
  __builtin_amdgcn_tensor_load_to_lds(g0, g1, z, z, 0);
#endif
}
#endif // HAVE_TDM

// ---------------------------------------------------------------------------
// Conv 4x4 stride-2 pad-1 + bias + ReLU as implicit GEMM.
// One 16(M)x16(N) tile per wave, K = Cin*16 in chunks of 32 (bf16 WMMA).
// Hw>=16: the 16 rows of a tile share one Cin x 4 x 34 input window ->
//   staged once into LDS (kills the 16x im2col re-fetch amplification).
// Weights (<=32KB fp32) staged per block via the Tensor Data Mover.
// ---------------------------------------------------------------------------
template<typename IT, int Cin, int Cout, int Hw>
__global__ __launch_bounds__(TPB) void conv_wmma_k(
    const IT* __restrict__ in, const float* __restrict__ w,
    const float* __restrict__ bias, bf16* __restrict__ out, int B)
{
  constexpr int Hin = Hw * 2;
  constexpr int K   = Cin * 16;
  constexpr int Kc  = (K + 31) / 32;
  constexpr int NT  = (Cout + 15) / 16;
  constexpr bool STAGE_A = (Hw >= 16);
  constexpr int  WN   = Cout * Cin * 16;         // weight element count
  constexpr bool WLDS = (WN * 4 <= 32768);       // conv0..conv3
  constexpr int  SW   = 36;                      // padded window row stride

  __shared__ float wbuf[WLDS ? WN : 1];
  __shared__ bf16  abuf[STAGE_A ? WPB * Cin * 4 * SW : 1];

  const int t     = blockIdx.y;
  const int lane  = threadIdx.x & 31;
  const int wslot = threadIdx.x >> 5;
  const int wid   = blockIdx.x * WPB + wslot;    // grids are exact: no guard
  const int nTile = wid % NT, mTile = wid / NT;

  const IT*    inT = in   + (size_t)t * B * Cin * Hin * Hin;
  const float* wT  = w    + (size_t)t * WN;
  const float* bT  = bias + (size_t)t * Cout;
  bf16*        oT  = out  + (size_t)t * B * Cout * Hw * Hw;

  const int n   = lane & 15;
  const int nG  = nTile * 16 + n;
  const int kbA = (lane >> 4) * 16;
  const int kbB = (lane >> 4) * 16;

  // ---- stage weights into LDS (TDM when available) ----
  if constexpr (WLDS) {
#if HAVE_TDM
    if (wslot == 0) tdm_load_f32(wbuf, wT, WN);
#else
    for (int i = threadIdx.x; i < WN; i += TPB) wbuf[i] = wT[i];
#endif
  }

  // ---- stage A input window into LDS (Hw >= 16) ----
  if constexpr (STAGE_A) {
    const int base = mTile * 16;
    const int bi   = base / (Hw * Hw);
    const int rem  = base - bi * (Hw * Hw);
    const int oh   = rem / Hw;
    const int owB  = rem - oh * Hw;              // multiple of 16
    bf16* ab = abuf + wslot * (Cin * 4 * SW);
    for (int idx = lane; idx < Cin * 4 * 34; idx += 32) {
      int ci = idx / (4 * 34);
      int r2 = idx - ci * (4 * 34);
      int rr = r2 / 34, cc = r2 - rr * 34;
      int ih = oh * 2 - 1 + rr, iw = owB * 2 - 1 + cc;
      float v = 0.f;
      if (ih >= 0 && ih < Hin && iw >= 0 && iw < Hin)
        v = ldf(inT + ((size_t)(bi * Cin + ci) * Hin + ih) * Hin + iw);
      ab[(ci * 4 + rr) * SW + cc] = (bf16)v;
    }
  }

  if constexpr (WLDS || STAGE_A) {
#if HAVE_TDM
    if constexpr (WLDS) {
      if (wslot == 0) __builtin_amdgcn_s_wait_tensorcnt(0);
    }
#endif
    __syncthreads();
  }

  // per-lane row decode for the direct (unstaged) A path
  int bi2 = 0, oh2 = 0, ow2 = 0;
  if constexpr (!STAGE_A) {
    const int mRow = mTile * 16 + (lane & 15);
    bi2 = mRow / (Hw * Hw);
    const int rem = mRow - bi2 * (Hw * Hw);
    oh2 = rem / Hw; ow2 = rem - oh2 * Hw;
  }

  v8f acc = {};
  for (int kc = 0; kc < Kc; ++kc) {
    v16bf a, bb;
#pragma unroll
    for (int e = 0; e < 16; ++e) {
      int k = kc * 32 + kbA + e;
      if constexpr (STAGE_A) {
        const bf16* ab = abuf + wslot * (Cin * 4 * SW);
        bf16 v = (bf16)0.f;
        if (k < K) {
          int ci = k >> 4, r = (k >> 2) & 3, s = k & 3;
          v = ab[(ci * 4 + r) * SW + 2 * (lane & 15) + s];
        }
        a[e] = v;
      } else {
        float v = 0.f;
        if (k < K) {
          int ci = k >> 4, r = (k >> 2) & 3, s = k & 3;
          int ih = oh2 * 2 - 1 + r, iw = ow2 * 2 - 1 + s;
          if (ih >= 0 && ih < Hin && iw >= 0 && iw < Hin)
            v = ldf(inT + ((size_t)(bi2 * Cin + ci) * Hin + ih) * Hin + iw);
        }
        a[e] = (bf16)v;
      }
    }
#pragma unroll
    for (int e = 0; e < 16; ++e) {
      int k = kc * 32 + kbB + e;
      float v = 0.f;
      if (k < K && nG < Cout) {
        int ci = k >> 4, rs = k & 15;
        if constexpr (WLDS) v = wbuf[(nG * Cin + ci) * 16 + rs];
        else                v = wT[(nG * Cin + ci) * 16 + rs];
      }
      bb[e] = (bf16)v;
    }
    acc = wmma_bf16(a, bb, acc);
  }

  if (nG < Cout) {
    float bv = bT[nG];
#pragma unroll
    for (int v = 0; v < 8; ++v) {
      int m  = mTile * 16 + v + ((lane >> 4) << 3);
      int b2 = m / (Hw * Hw), r2 = m - b2 * (Hw * Hw);
      int oh3 = r2 / Hw, ow3 = r2 - oh3 * Hw;
      float x = acc[v] + bv; x = x > 0.f ? x : 0.f;
      oT[((size_t)(b2 * Cout + nG) * Hw + oh3) * Hw + ow3] = (bf16)x;
    }
  }
}

// ---------------------------------------------------------------------------
// Fused MLP 128 -> 32 -> 32 -> 64. Weights TDM-staged in LDS (28 KB, shared
// by all 8 waves); inter-layer tiles bounce through LDS (C-layout -> A-layout).
// ---------------------------------------------------------------------------
__global__ __launch_bounds__(TPB) void mlp_k(
    const bf16* __restrict__ enc,                 // [6144][128] bf16
    const float* __restrict__ w1, const float* __restrict__ b1,
    const float* __restrict__ w2, const float* __restrict__ b2,
    const float* __restrict__ w3, const float* __restrict__ b3,
    float* __restrict__ feats)                    // [6144][64] f32
{
  __shared__ bf16  hbuf[WPB][16 * 32];
  __shared__ float w1L[32 * 128];
  __shared__ float w2L[32 * 32];
  __shared__ float w3L[64 * 32];
  const int lane  = threadIdx.x & 31;
  const int ws    = threadIdx.x >> 5;
  const int mTile = blockIdx.x * WPB + ws;
  const int n   = lane & 15;
  const int kbA = (lane >> 4) * 16;
  const int kbB = (lane >> 4) * 16;

#if HAVE_TDM
  if (ws == 0) {
    tdm_load_f32(w1L, w1, 32 * 128);
    tdm_load_f32(w2L, w2, 32 * 32);
    tdm_load_f32(w3L, w3, 64 * 32);
    __builtin_amdgcn_s_wait_tensorcnt(0);
  }
#else
  for (int i = threadIdx.x; i < 32 * 128; i += TPB) w1L[i] = w1[i];
  for (int i = threadIdx.x; i < 32 * 32;  i += TPB) w2L[i] = w2[i];
  for (int i = threadIdx.x; i < 64 * 32;  i += TPB) w3L[i] = w3[i];
#endif
  __syncthreads();

  // layer 1: 128 -> 32
  v8f acc[2] = {};
  for (int kc = 0; kc < 4; ++kc) {
    v16bf a;
#pragma unroll
    for (int e = 0; e < 16; ++e)
      a[e] = enc[(size_t)(mTile * 16 + (lane & 15)) * 128 + kc * 32 + kbA +
                 e];
#pragma unroll
    for (int nt = 0; nt < 2; ++nt) {
      v16bf bb;
#pragma unroll
      for (int e = 0; e < 16; ++e)
        bb[e] = (bf16)w1L[(nt * 16 + n) * 128 + kc * 32 + kbB + e];
      acc[nt] = wmma_bf16(a, bb, acc[nt]);
    }
  }
#pragma unroll
  for (int nt = 0; nt < 2; ++nt) {
    float bv = b1[nt * 16 + n];
#pragma unroll
    for (int v = 0; v < 8; ++v) {
      int row = v + ((lane >> 4) << 3);
      float x = acc[nt][v] + bv; x = x > 0.f ? x : 0.f;
      hbuf[ws][row * 32 + nt * 16 + n] = (bf16)x;
    }
  }
  __syncthreads();

  // layer 2: 32 -> 32
  {
    v16bf a;
#pragma unroll
    for (int e = 0; e < 16; ++e)
      a[e] = hbuf[ws][(lane & 15) * 32 + kbA + e];
    v8f a2[2] = {};
#pragma unroll
    for (int nt = 0; nt < 2; ++nt) {
      v16bf bb;
#pragma unroll
      for (int e = 0; e < 16; ++e)
        bb[e] = (bf16)w2L[(nt * 16 + n) * 32 + kbB + e];
      a2[nt] = wmma_bf16(a, bb, a2[nt]);
    }
    __syncthreads();
#pragma unroll
    for (int nt = 0; nt < 2; ++nt) {
      float bv = b2[nt * 16 + n];
#pragma unroll
      for (int v = 0; v < 8; ++v) {
        int row = v + ((lane >> 4) << 3);
        float x = a2[nt][v] + bv; x = x > 0.f ? x : 0.f;
        hbuf[ws][row * 32 + nt * 16 + n] = (bf16)x;
      }
    }
  }
  __syncthreads();

  // layer 3: 32 -> 64, linear
  {
    v16bf a;
#pragma unroll
    for (int e = 0; e < 16; ++e)
      a[e] = hbuf[ws][(lane & 15) * 32 + kbA + e];
#pragma unroll
    for (int nt = 0; nt < 4; ++nt) {
      v16bf bb;
#pragma unroll
      for (int e = 0; e < 16; ++e)
        bb[e] = (bf16)w3L[(nt * 16 + n) * 32 + kbB + e];
      v8f z = {};
      v8f a3 = wmma_bf16(a, bb, z);
      float bv = b3[nt * 16 + n];
#pragma unroll
      for (int v = 0; v < 8; ++v) {
        int row = mTile * 16 + v + ((lane >> 4) << 3);
        feats[(size_t)row * 64 + nt * 16 + n] = a3[v] + bv;
      }
    }
  }
}

// ---------------------------------------------------------------------------
// LSTM(64 -> 20), T=3. One wave owns 16 batch rows across all time steps.
// w_ih / w_hh TDM-staged once (26 KB); h/c state in LDS per wave.
// ---------------------------------------------------------------------------
__global__ __launch_bounds__(TPB) void lstm_k(
    const float* __restrict__ feats,   // [3*B][64]
    const float* __restrict__ wih,     // [80][64]
    const float* __restrict__ whh,     // [80][20]
    const float* __restrict__ bih, const float* __restrict__ bhh,
    float* __restrict__ hseq, int B)   // [3*B][20]
{
  __shared__ float gbuf[WPB][16 * 80];
  __shared__ bf16  hbuf[WPB][16 * 32];
  __shared__ float cbuf[WPB][16 * 20];
  __shared__ float wihL[80 * 64];
  __shared__ float whhL[80 * 20];
  const int lane = threadIdx.x & 31;
  const int ws   = threadIdx.x >> 5;
  const int rowB = (blockIdx.x * WPB + ws) * 16;
  const int n    = lane & 15;
  const int kbA  = (lane >> 4) * 16;
  const int kbB  = (lane >> 4) * 16;

#if HAVE_TDM
  if (ws == 0) {
    tdm_load_f32(wihL, wih, 80 * 64);
    tdm_load_f32(whhL, whh, 80 * 20);
    __builtin_amdgcn_s_wait_tensorcnt(0);
  }
#else
  for (int i = threadIdx.x; i < 80 * 64; i += TPB) wihL[i] = wih[i];
  for (int i = threadIdx.x; i < 80 * 20; i += TPB) whhL[i] = whh[i];
#endif
  for (int i = lane; i < 16 * 32; i += 32) hbuf[ws][i] = (bf16)0.f;
  for (int i = lane; i < 16 * 20; i += 32) cbuf[ws][i] = 0.f;
  __syncthreads();

  for (int t = 0; t < 3; ++t) {
    v16bf ax0, ax1, ah;
    const float* xr = feats + (size_t)(t * B + rowB + (lane & 15)) * 64;
#pragma unroll
    for (int e = 0; e < 16; ++e) {
      int ko = kbA + e;
      ax0[e] = (bf16)xr[ko];
      ax1[e] = (bf16)xr[32 + ko];
      ah[e]  = hbuf[ws][(lane & 15) * 32 + ko];
    }
#pragma unroll
    for (int nt = 0; nt < 5; ++nt) {
      int gn = nt * 16 + n;
      v16bf b0, b1v, bh;
#pragma unroll
      for (int e = 0; e < 16; ++e) {
        b0[e]  = (bf16)wihL[gn * 64 + kbB + e];
        b1v[e] = (bf16)wihL[gn * 64 + 32 + kbB + e];
        int k = kbB + e;
        bh[e]  = (bf16)(k < 20 ? whhL[gn * 20 + k] : 0.f);
      }
      v8f acc = {};
      acc = wmma_bf16(ax0, b0,  acc);
      acc = wmma_bf16(ax1, b1v, acc);
      acc = wmma_bf16(ah,  bh,  acc);
      float bv = bih[gn] + bhh[gn];
#pragma unroll
      for (int v = 0; v < 8; ++v)
        gbuf[ws][(v + ((lane >> 4) << 3)) * 80 + gn] = acc[v] + bv;
    }
    __syncthreads();
    for (int idx = lane; idx < 16 * 20; idx += 32) {
      int row = idx / 20, j = idx - (idx / 20) * 20;
      float gi = gbuf[ws][row * 80 + j];
      float gf = gbuf[ws][row * 80 + 20 + j];
      float gg = gbuf[ws][row * 80 + 40 + j];
      float go = gbuf[ws][row * 80 + 60 + j];
      float c = 1.f / (1.f + expf(-gf)) * cbuf[ws][idx] +
                1.f / (1.f + expf(-gi)) * tanhf(gg);
      float h = 1.f / (1.f + expf(-go)) * tanhf(c);
      cbuf[ws][idx] = c;
      hbuf[ws][row * 32 + j] = (bf16)h;
      hseq[(size_t)(t * B + rowB + row) * 20 + j] = h;
    }
    __syncthreads();
  }
}

// ---------------------------------------------------------------------------
// 20 reward heads: 20 -> 32 -> 32 -> 1.  One wave = (head, 16 rows).
// ---------------------------------------------------------------------------
__global__ __launch_bounds__(TPB) void heads_k(
    const float* __restrict__ hseq,    // [6144][20]
    const float* __restrict__ w1, const float* __restrict__ b1,
    const float* __restrict__ w2, const float* __restrict__ b2,
    const float* __restrict__ w3, const float* __restrict__ b3,
    float* __restrict__ out, int B)    // [20][3][B]
{
  __shared__ bf16 xbuf[WPB][16 * 32];
  const int lane = threadIdx.x & 31;
  const int ws   = threadIdx.x >> 5;
  const int wid  = blockIdx.x * WPB + ws;
  const int MT   = (3 * B) / 16;
  const int kh   = wid / MT, mTile = wid % MT;
  const int n    = lane & 15;
  const int kbA  = (lane >> 4) * 16, kbB = (lane >> 4) * 16;
  const float* W1 = w1 + kh * 32 * 20;
  const float* W2 = w2 + kh * 32 * 32;
  const float* W3 = w3 + kh * 32;

  // layer 1: K=20 (padded to 32) -> 32, relu
  v16bf a;
  {
    const float* hr = hseq + (size_t)(mTile * 16 + (lane & 15)) * 20;
#pragma unroll
    for (int e = 0; e < 16; ++e) {
      int k = kbA + e;
      a[e] = (bf16)(k < 20 ? hr[k] : 0.f);
    }
  }
#pragma unroll
  for (int nt = 0; nt < 2; ++nt) {
    v16bf bb;
#pragma unroll
    for (int e = 0; e < 16; ++e) {
      int k = kbB + e;
      bb[e] = (bf16)(k < 20 ? W1[(nt * 16 + n) * 20 + k] : 0.f);
    }
    v8f z = {};
    v8f acc = wmma_bf16(a, bb, z);
    float bv = b1[kh * 32 + nt * 16 + n];
#pragma unroll
    for (int v = 0; v < 8; ++v) {
      int row = v + ((lane >> 4) << 3);
      float x = acc[v] + bv; x = x > 0.f ? x : 0.f;
      xbuf[ws][row * 32 + nt * 16 + n] = (bf16)x;
    }
  }
  __syncthreads();

  // layer 2: 32 -> 32, relu
  {
    v16bf a2;
#pragma unroll
    for (int e = 0; e < 16; ++e)
      a2[e] = xbuf[ws][(lane & 15) * 32 + kbA + e];
    v8f acc2[2] = {};
#pragma unroll
    for (int nt = 0; nt < 2; ++nt) {
      v16bf bb;
#pragma unroll
      for (int e = 0; e < 16; ++e)
        bb[e] = (bf16)W2[(nt * 16 + n) * 32 + kbB + e];
      acc2[nt] = wmma_bf16(a2, bb, acc2[nt]);
    }
    __syncthreads();
#pragma unroll
    for (int nt = 0; nt < 2; ++nt) {
      float bv = b2[kh * 32 + nt * 16 + n];
#pragma unroll
      for (int v = 0; v < 8; ++v) {
        int row = v + ((lane >> 4) << 3);
        float x = acc2[nt][v] + bv; x = x > 0.f ? x : 0.f;
        xbuf[ws][row * 32 + nt * 16 + n] = (bf16)x;
      }
    }
  }
  __syncthreads();

  // layer 3: 32 -> 1 (scalar dot per row, lanes 0..15)
  if (lane < 16) {
    float s = b3[kh];
#pragma unroll
    for (int j = 0; j < 32; ++j)
      s += (float)xbuf[ws][lane * 32 + j] * W3[j];
    int m = mTile * 16 + lane;
    int t = m / B, bi = m - t * B;
    out[((size_t)kh * 3 + t) * B + bi] = s;
  }
}

// ---------------------------------------------------------------------------
extern "C" void kernel_launch(void* const* d_in, const int* in_sizes, int n_in,
                              void* d_out, int out_size, void* d_ws, size_t ws_size,
                              hipStream_t stream)
{
  (void)in_sizes; (void)n_in; (void)out_size; (void)ws_size;
  const float* img = (const float*)d_in[0];
  const float* ew[6]; const float* eb[6];
  for (int l = 0; l < 6; ++l) { ew[l] = (const float*)d_in[1 + 2 * l];
                                eb[l] = (const float*)d_in[2 + 2 * l]; }
  const float* mw1 = (const float*)d_in[13]; const float* mb1 = (const float*)d_in[14];
  const float* mw2 = (const float*)d_in[15]; const float* mb2 = (const float*)d_in[16];
  const float* mw3 = (const float*)d_in[17]; const float* mb3 = (const float*)d_in[18];
  const float* wih = (const float*)d_in[19]; const float* whh = (const float*)d_in[20];
  const float* bih = (const float*)d_in[21]; const float* bhh = (const float*)d_in[22];
  const float* hw1 = (const float*)d_in[23]; const float* hb1 = (const float*)d_in[24];
  const float* hw2 = (const float*)d_in[25]; const float* hb2 = (const float*)d_in[26];
  const float* hw3 = (const float*)d_in[27]; const float* hb3 = (const float*)d_in[28];
  float* out = (float*)d_out;

  const int T = 3, B = 2048;
  char* ws = (char*)d_ws;
  bf16*  A0 = (bf16*)(ws);                 // 3*2048*4*32*32 bf16
  bf16*  A1 = (bf16*)(ws + 50331648);      // 3*2048*8*16*16
  bf16*  A2 = (bf16*)(ws + 75497472);      // 3*2048*16*8*8
  bf16*  A3 = (bf16*)(ws + 88080384);      // 3*2048*32*4*4
  bf16*  A4 = (bf16*)(ws + 94371840);      // 3*2048*64*2*2
  bf16*  A5 = (bf16*)(ws + 97517568);      // 3*2048*128 (enc)
  float* feats = (float*)(ws + 99090432);  // 3*2048*64 f32
  float* hseq  = (float*)(ws + 100663296); // 3*2048*20 f32

  dim3 blk(TPB, 1, 1);
  auto g = [](int M, int NT) { return ((M / 16) * NT + WPB - 1) / WPB; };

  conv_wmma_k<float, 3,   4, 32><<<dim3(g(B*32*32, 1), T, 1), blk, 0, stream>>>(img, ew[0], eb[0], A0, B);
  conv_wmma_k<bf16,  4,   8, 16><<<dim3(g(B*16*16, 1), T, 1), blk, 0, stream>>>(A0,  ew[1], eb[1], A1, B);
  conv_wmma_k<bf16,  8,  16,  8><<<dim3(g(B*8*8,   1), T, 1), blk, 0, stream>>>(A1,  ew[2], eb[2], A2, B);
  conv_wmma_k<bf16, 16,  32,  4><<<dim3(g(B*4*4,   2), T, 1), blk, 0, stream>>>(A2,  ew[3], eb[3], A3, B);
  conv_wmma_k<bf16, 32,  64,  2><<<dim3(g(B*2*2,   4), T, 1), blk, 0, stream>>>(A3,  ew[4], eb[4], A4, B);
  conv_wmma_k<bf16, 64, 128,  1><<<dim3(g(B*1*1,   8), T, 1), blk, 0, stream>>>(A4,  ew[5], eb[5], A5, B);

  mlp_k  <<<dim3((T * B / 16) / WPB, 1, 1), blk, 0, stream>>>(A5, mw1, mb1, mw2, mb2, mw3, mb3, feats);
  lstm_k <<<dim3((B / 16) / WPB, 1, 1),     blk, 0, stream>>>(feats, wih, whh, bih, bhh, hseq, B);
  heads_k<<<dim3((20 * (T * B / 16)) / WPB, 1, 1), blk, 0, stream>>>(hseq, hw1, hb1, hw2, hb2, hw3, hb3, out, B);
}
